// AttentionPooling_21973052686567
// MI455X (gfx1250) — compile-verified, matrix operand order checked by reading, not run on hardware
//
#include <hip/hip_runtime.h>
#include <hip/hip_bf16.h>

typedef _Float16 v16h __attribute__((ext_vector_type(16)));
typedef _Float16 v8h  __attribute__((ext_vector_type(8)));
typedef float    v8f  __attribute__((ext_vector_type(8)));

// ---------------- order-preserving float <-> u32 key (for atomic max) -------
__device__ __forceinline__ unsigned fkey(float f) {
    unsigned u = __float_as_uint(f);
    return (u & 0x80000000u) ? ~u : (u | 0x80000000u);
}
__device__ __forceinline__ float funkey(unsigned k) {
    return __uint_as_float((k & 0x80000000u) ? (k & 0x7fffffffu) : ~k);
}

// ---------------- kernel 1: scores = LReLU(x@W1 + b1) @ W2 + b2 (WMMA) ------
// block = 256 threads = 8 waves; each wave computes one 16-row tile.
__global__ __launch_bounds__(256) void scores_kernel(
    const float* __restrict__ x, const float* __restrict__ W1,
    const float* __restrict__ b1, const float* __restrict__ W2,
    const float* __restrict__ b2, float* __restrict__ scores, int N) {
    __shared__ _Float16 ldsW[64 * 128];  // W1^T as f16: [h][d], row stride 128

    // stage W1^T (contiguous LDS stores; global reads are tiny and L2-resident)
    for (int i = threadIdx.x; i < 64 * 128; i += 256) {
        int h = i >> 7, d = i & 127;
        ldsW[i] = (_Float16)W1[d * 64 + h];
    }
    __syncthreads();

    const int wave = threadIdx.x >> 5;
    const int lane = threadIdx.x & 31;
    const int row0 = (blockIdx.x * 8 + wave) * 16;
    if (row0 + 16 > N) return;

    const int lr = lane & 15;   // row/col within half-wave group
    const int lh = lane >> 4;   // which half-wave

    v8f acc[4] = {};            // 4 N-tiles of 16 (H = 64)

    // A row for this lane (16-bit A layout: lane -> row lr, K chunks per lh)
    const float* __restrict__ xrow = x + (long)(row0 + lr) * 128;

    for (int kt = 0; kt < 4; ++kt) {           // K = 128 in 4 tiles of 32
        const int c0 = kt * 32 + lh * 8;       // first 8-elem K chunk
        const float4* __restrict__ p = (const float4*)(xrow + c0);
        float4 u0 = p[0], u1 = p[1];           // K = c0 .. c0+7
        float4 u2 = p[4], u3 = p[5];           // K = c0+16 .. c0+23
        v16h a;
        a[0] = (_Float16)u0.x;  a[1] = (_Float16)u0.y;
        a[2] = (_Float16)u0.z;  a[3] = (_Float16)u0.w;
        a[4] = (_Float16)u1.x;  a[5] = (_Float16)u1.y;
        a[6] = (_Float16)u1.z;  a[7] = (_Float16)u1.w;
        a[8]  = (_Float16)u2.x; a[9]  = (_Float16)u2.y;
        a[10] = (_Float16)u2.z; a[11] = (_Float16)u2.w;
        a[12] = (_Float16)u3.x; a[13] = (_Float16)u3.y;
        a[14] = (_Float16)u3.z; a[15] = (_Float16)u3.w;

#pragma unroll
        for (int nt = 0; nt < 4; ++nt) {       // N = 64 in 4 tiles of 16
            const _Float16* wp = &ldsW[(nt * 16 + lr) * 128 + c0];
            v8h blo = *(const v8h*)(wp);        // 16B-aligned LDS b128
            v8h bhi = *(const v8h*)(wp + 16);
            v16h bm;
#pragma unroll
            for (int j = 0; j < 8; ++j) { bm[j] = blo[j]; bm[8 + j] = bhi[j]; }
            acc[nt] = __builtin_amdgcn_wmma_f32_16x16x32_f16(
                false, a, false, bm, (short)0, acc[nt], false, false);
        }
    }

    // epilogue: + b1, LeakyReLU, dot with W2 across N, + b2
    float partial[8];
#pragma unroll
    for (int r = 0; r < 8; ++r) partial[r] = 0.f;
#pragma unroll
    for (int nt = 0; nt < 4; ++nt) {
        const float bb = b1[nt * 16 + lr];
        const float ww = W2[nt * 16 + lr];
#pragma unroll
        for (int r = 0; r < 8; ++r) {
            float hv = acc[nt][r] + bb;
            hv = (hv >= 0.f) ? hv : 0.01f * hv;
            partial[r] += hv * ww;
        }
    }
    const float bias2 = b2[0];
#pragma unroll
    for (int r = 0; r < 8; ++r) {
        float v = partial[r];             // reduce over 16 lanes of same row
        v += __shfl_xor(v, 1);
        v += __shfl_xor(v, 2);
        v += __shfl_xor(v, 4);
        v += __shfl_xor(v, 8);
        if (lr == 0) scores[row0 + r + 8 * lh] = v + bias2;  // C layout: M=r / M=8+r
    }
}

// ---------------- kernel 2: segment max (keyed u32 atomicMax) ---------------
__global__ __launch_bounds__(256) void segmax_kernel(
    const float* __restrict__ scores, const int* __restrict__ batch,
    unsigned* __restrict__ segmax, int N) {
    int i = blockIdx.x * 256 + threadIdx.x;
    if (i >= N) return;
    atomicMax(&segmax[batch[i]], fkey(scores[i]));
}

// ---------------- kernel 3: e = exp(s - max), segment sum -------------------
__global__ __launch_bounds__(256) void expsum_kernel(
    float* __restrict__ scores, const int* __restrict__ batch,
    const unsigned* __restrict__ segmax, float* __restrict__ segsum, int N) {
    int i = blockIdx.x * 256 + threadIdx.x;
    if (i >= N) return;
    int b = batch[i];
    float e = __expf(scores[i] - funkey(segmax[b]));
    scores[i] = e;                       // reuse buffer: scores -> e
    atomicAdd(&segsum[b], e);
}

// ---------------- kernel 4: weighted pooling (sorted-run accumulation) ------
// block = 256 threads = 8 waves; each wave owns 128 consecutive nodes,
// lane owns 4 dims (float4) -> one fully coalesced 512B row read per node.
__global__ __launch_bounds__(256) void pool_kernel(
    const float* __restrict__ x, const int* __restrict__ batch,
    const float* __restrict__ e, const float* __restrict__ segsum,
    float* __restrict__ out, int N) {
    const int wave = threadIdx.x >> 5;
    const int lane = threadIdx.x & 31;
    const long gw = (long)blockIdx.x * 8 + wave;
    long n0 = gw * 128;
    if (n0 >= N) return;
    long n1 = n0 + 128; if (n1 > N) n1 = N;
    const int d0 = lane * 4;

    float4 acc = {0.f, 0.f, 0.f, 0.f};
    int cur = batch[n0];
    for (long n = n0; n < n1; ++n) {
        int b = batch[n];
        if (b != cur) {
            float* o = out + (long)cur * 128 + d0;
            atomicAdd(o + 0, acc.x); atomicAdd(o + 1, acc.y);
            atomicAdd(o + 2, acc.z); atomicAdd(o + 3, acc.w);
            acc = {0.f, 0.f, 0.f, 0.f};
            cur = b;
        }
        float a = e[n] / (segsum[b] + 1e-16f);
        float4 xv = ((const float4*)(x + n * 128))[lane];
        acc.x += xv.x * a; acc.y += xv.y * a;
        acc.z += xv.z * a; acc.w += xv.w * a;
    }
    float* o = out + (long)cur * 128 + d0;
    atomicAdd(o + 0, acc.x); atomicAdd(o + 1, acc.y);
    atomicAdd(o + 2, acc.z); atomicAdd(o + 3, acc.w);
}

// ---------------------------------------------------------------------------
extern "C" void kernel_launch(void* const* d_in, const int* in_sizes, int n_in,
                              void* d_out, int out_size, void* d_ws, size_t ws_size,
                              hipStream_t stream) {
    const float* x     = (const float*)d_in[0];
    const int*   batch = (const int*)  d_in[1];
    const float* W1    = (const float*)d_in[2];
    const float* b1    = (const float*)d_in[3];
    const float* W2    = (const float*)d_in[4];
    const float* b2    = (const float*)d_in[5];
    float*       out   = (float*)d_out;

    const int N = in_sizes[1];          // number of nodes
    const int G = out_size / 128;       // number of segments (D = 128)

    // workspace layout: scores/e [N] f32 | segmax keys [G] u32 | segsum [G] f32
    float*    scores = (float*)d_ws;
    unsigned* segmax = (unsigned*)(scores + N);
    float*    segsum = (float*)(segmax + G);

    // zero-init (key 0 == minimal monotonic float key); graph-capturable memsets
    (void)hipMemsetAsync(out, 0, (size_t)out_size * sizeof(float), stream);
    (void)hipMemsetAsync(segmax, 0, (size_t)G * 2 * sizeof(unsigned), stream);

    const int blocks1 = (N + 127) / 128;           // 8 waves x 16 rows per block
    scores_kernel<<<blocks1, 256, 0, stream>>>(x, W1, b1, W2, b2, scores, N);

    const int blocksN = (N + 255) / 256;
    segmax_kernel<<<blocksN, 256, 0, stream>>>(scores, batch, segmax, N);
    expsum_kernel<<<blocksN, 256, 0, stream>>>(scores, batch, segmax, segsum, N);

    const long waves = ((long)N + 127) / 128;
    const int blocks4 = (int)((waves + 7) / 8);
    pool_kernel<<<blocks4, 256, 0, stream>>>(x, batch, scores, segsum, out, N);
}
